// MultiHeadAttention_23270132810335
// MI455X (gfx1250) — matrix operand order repaired and checked by reading, not compiled, not run on hardware
//
#include <hip/hip_runtime.h>

typedef __bf16 bf16;
typedef __attribute__((ext_vector_type(16))) __bf16 v16b;
typedef __attribute__((ext_vector_type(8)))  __bf16 v8b;
typedef __attribute__((ext_vector_type(8)))  float  v8f;
typedef __attribute__((ext_vector_type(4)))  unsigned int v4u;
typedef __attribute__((ext_vector_type(8)))  int v8i;
typedef __attribute__((ext_vector_type(4)))  int v4i;

#if __has_builtin(__builtin_amdgcn_tensor_load_to_lds) && __has_builtin(__builtin_amdgcn_s_wait_tensorcnt)
#define HAVE_TDM 1
#else
#define HAVE_TDM 0
#endif

// ---------------------------------------------------------------------------
// fp32 -> bf16 conversion (grid-stride)
// ---------------------------------------------------------------------------
__global__ void __launch_bounds__(256)
f32_to_bf16_kernel(const float* __restrict__ in, bf16* __restrict__ out, size_t n)
{
    size_t i = (size_t)blockIdx.x * blockDim.x + threadIdx.x;
    size_t stride = (size_t)gridDim.x * blockDim.x;
    for (; i < n; i += stride) out[i] = (bf16)in[i];
}

// ---------------------------------------------------------------------------
// WMMA fragment loaders (contiguous k order: k = 16*(lane/16) + s, wave32)
// ---------------------------------------------------------------------------
__device__ __forceinline__ v16b load_a_frag(const bf16* __restrict__ A, int lda,
                                            int m0, int k0, int lane)
{
    int row = lane & 15;
    int kb  = k0 + ((lane >> 4) << 4);           // +0 or +16
    const bf16* p = A + (size_t)(m0 + row) * lda + kb;
    union { v16b v; v8b h[2]; } u;
    u.h[0] = *reinterpret_cast<const v8b*>(p);        // K = kb .. kb+7
    u.h[1] = *reinterpret_cast<const v8b*>(p + 8);    // K = kb+8 .. kb+15
    return u.v;
}

// B 32x16 (KxN) sourced from row-major [N,K] global storage
__device__ __forceinline__ v16b load_b_frag(const bf16* __restrict__ B, int ldb,
                                            int n0, int k0, int lane)
{
    int col = lane & 15;
    int kb  = k0 + ((lane >> 4) << 4);           // +0 or +16
    return *reinterpret_cast<const v16b*>(B + (size_t)(n0 + col) * ldb + kb);
}

// B fragment from an LDS-resident 64x32 bf16 tile (row-major, rows of 32 elems)
__device__ __forceinline__ v16b load_b_frag_lds(const bf16* Bl, int t, int lane)
{
    int col = lane & 15;
    int kb  = (lane >> 4) << 4;                  // +0 or +16
    return *reinterpret_cast<const v16b*>(Bl + (size_t)(t * 16 + col) * 32 + kb);
}

#if HAVE_TDM
// ---------------------------------------------------------------------------
// TDM: async-load a 64(rows) x 32(K) bf16 tile from global (row stride = ldb
// elements) into LDS at byte offset lds_off. D# per CDNA5 ISA ch.8.
// Tracked with TENSORcnt (s_wait_tensorcnt).
// ---------------------------------------------------------------------------
__device__ __forceinline__ void tdm_load_tile_64x32(const bf16* gsrc, int ldb_elems,
                                                    unsigned lds_off)
{
    unsigned long ga = (unsigned long)(const void*)gsrc;
    v4u g0;
    g0[0] = 1u;                                           // count=1, user-mode
    g0[1] = lds_off;                                      // lds_addr (bytes)
    g0[2] = (unsigned)(ga & 0xffffffffu);                 // global_addr[31:0]
    g0[3] = (unsigned)((ga >> 32) & 0x01ffffffu)          // global_addr[56:32]
          | (2u << 30);                                   // type = 2 ("image")
    const unsigned td0 = 0x7fffffffu, td1 = 0x7fffffffu;  // huge tensor dims
    const unsigned long s0 = (unsigned long)(unsigned)ldb_elems;
    v8i g1;
    g1[0] = (int)(1u << 16);                              // data_size=2B, mask=0
    g1[1] = (int)((td0 & 0xffffu) << 16);                 // abar=0 | td0[15:0]
    g1[2] = (int)(((td0 >> 16) & 0xffffu) | ((td1 & 0xffffu) << 16));
    g1[3] = (int)(((td1 >> 16) & 0xffffu) | (32u << 16)); // tile_dim0 = 32
    g1[4] = 64;                                           // tile_dim1=64, tile_dim2=0
    g1[5] = (int)(s0 & 0xffffffffu);                      // dim0_stride[31:0]
    g1[6] = (int)((s0 >> 32) & 0xffffu);                  // dim0_stride[47:32]
    g1[7] = 0;                                            // dim1_stride unused
    v4i z4 = {0, 0, 0, 0};
#if defined(__clang_major__) && (__clang_major__ >= 23)
    v8i z8 = {0, 0, 0, 0, 0, 0, 0, 0};
    __builtin_amdgcn_tensor_load_to_lds(g0, g1, z4, z4, z8, 0);
#else
    __builtin_amdgcn_tensor_load_to_lds(g0, g1, z4, z4, 0);
#endif
}
#endif // HAVE_TDM

// One K=32 step: 2 A fragments (global), 4 B fragments (LDS tile), 8 WMMAs.
__device__ __forceinline__ void gemm_step_lds(const bf16* __restrict__ A, int lda,
                                              int m0, int k, int lane,
                                              const bf16* Bl, v8f (&acc)[2][4])
{
    v16b a0 = load_a_frag(A, lda, m0,      k, lane);
    v16b a1 = load_a_frag(A, lda, m0 + 16, k, lane);
    v16b b0 = load_b_frag_lds(Bl, 0, lane);
    v16b b1 = load_b_frag_lds(Bl, 1, lane);
    v16b b2 = load_b_frag_lds(Bl, 2, lane);
    v16b b3 = load_b_frag_lds(Bl, 3, lane);
    acc[0][0] = __builtin_amdgcn_wmma_f32_16x16x32_bf16(false, a0, false, b0, (short)0, acc[0][0], false, false);
    acc[0][1] = __builtin_amdgcn_wmma_f32_16x16x32_bf16(false, a0, false, b1, (short)0, acc[0][1], false, false);
    acc[0][2] = __builtin_amdgcn_wmma_f32_16x16x32_bf16(false, a0, false, b2, (short)0, acc[0][2], false, false);
    acc[0][3] = __builtin_amdgcn_wmma_f32_16x16x32_bf16(false, a0, false, b3, (short)0, acc[0][3], false, false);
    acc[1][0] = __builtin_amdgcn_wmma_f32_16x16x32_bf16(false, a1, false, b0, (short)0, acc[1][0], false, false);
    acc[1][1] = __builtin_amdgcn_wmma_f32_16x16x32_bf16(false, a1, false, b1, (short)0, acc[1][1], false, false);
    acc[1][2] = __builtin_amdgcn_wmma_f32_16x16x32_bf16(false, a1, false, b2, (short)0, acc[1][2], false, false);
    acc[1][3] = __builtin_amdgcn_wmma_f32_16x16x32_bf16(false, a1, false, b3, (short)0, acc[1][3], false, false);
}

// ---------------------------------------------------------------------------
// Batched NT GEMM: C[z] = scale * A[z] (MxK row-major) * B[z]^T (B NxK row-major)
// Per-z offsets: off = (z>>zshift)*X0 + (z&((1<<zshift)-1))*X1 for A, B, C.
// Block = 128 threads (4 waves); each wave computes a 32Mx64N tile
// (8 accumulators, 8 WMMAs per K=32 step). B tile staged through LDS by the
// Tensor Data Mover, double-buffered per wave on TENSORcnt; last K-step peeled
// so the steady-state loop is branch-free.
// ---------------------------------------------------------------------------
template<bool OUT_BF16, bool TRANS_C>
__global__ void __launch_bounds__(128)
gemm_nt_wmma(const bf16* __restrict__ Ab, const bf16* __restrict__ Bb,
             void* __restrict__ Cv,
             int K, int lda, int ldb, int ldc, int zshift,
             long A0, long A1, long B0, long B1, long C0, long C1,
             float scale)
{
    const int z  = blockIdx.z;
    const int zq = z >> zshift;
    const int zr = z & ((1 << zshift) - 1);
    const bf16* A = Ab + (size_t)zq * A0 + (size_t)zr * A1;
    const bf16* B = Bb + (size_t)zq * B0 + (size_t)zr * B1;
    const size_t coff = (size_t)zq * C0 + (size_t)zr * C1;

    const int lane = threadIdx.x & 31;
    const int wave = threadIdx.x >> 5;

    const int m0 = blockIdx.y * 128 + wave * 32;
    const int n0 = blockIdx.x * 64;

    v8f acc[2][4] = {};

#if HAVE_TDM
    __shared__ bf16 ldsB[4][2][64 * 32];                 // 32 KB: per-wave double buffer
    bf16* buf0 = &ldsB[wave][0][0];
    bf16* buf1 = &ldsB[wave][1][0];
    const unsigned off0 = (unsigned)(uintptr_t)buf0;     // generic LDS addr low 32b = offset
    const unsigned off1 = (unsigned)(uintptr_t)buf1;
    const bf16* Brow = B + (size_t)n0 * ldb;

    tdm_load_tile_64x32(Brow, ldb, off0);

    int k = 0;
    int cur = 0;
    // steady state: unconditional prefetch of next tile, wait for current
    for (; k < K - 32; k += 32) {
        tdm_load_tile_64x32(Brow + (k + 32), ldb, cur ? off0 : off1);
        __builtin_amdgcn_s_wait_tensorcnt(1);            // current buffer resident
        gemm_step_lds(A, lda, m0, k, lane, cur ? buf1 : buf0, acc);
        cur ^= 1;
    }
    // peeled tail: no prefetch, drain TENSORcnt
    __builtin_amdgcn_s_wait_tensorcnt(0);
    gemm_step_lds(A, lda, m0, k, lane, cur ? buf1 : buf0, acc);
#else
    for (int k = 0; k < K; k += 32) {
        v16b a0 = load_a_frag(A, lda, m0,      k, lane);
        v16b a1 = load_a_frag(A, lda, m0 + 16, k, lane);
        v16b b0 = load_b_frag(B, ldb, n0 +  0, k, lane);
        v16b b1 = load_b_frag(B, ldb, n0 + 16, k, lane);
        v16b b2 = load_b_frag(B, ldb, n0 + 32, k, lane);
        v16b b3 = load_b_frag(B, ldb, n0 + 48, k, lane);
        acc[0][0] = __builtin_amdgcn_wmma_f32_16x16x32_bf16(false, a0, false, b0, (short)0, acc[0][0], false, false);
        acc[0][1] = __builtin_amdgcn_wmma_f32_16x16x32_bf16(false, a0, false, b1, (short)0, acc[0][1], false, false);
        acc[0][2] = __builtin_amdgcn_wmma_f32_16x16x32_bf16(false, a0, false, b2, (short)0, acc[0][2], false, false);
        acc[0][3] = __builtin_amdgcn_wmma_f32_16x16x32_bf16(false, a0, false, b3, (short)0, acc[0][3], false, false);
        acc[1][0] = __builtin_amdgcn_wmma_f32_16x16x32_bf16(false, a1, false, b0, (short)0, acc[1][0], false, false);
        acc[1][1] = __builtin_amdgcn_wmma_f32_16x16x32_bf16(false, a1, false, b1, (short)0, acc[1][1], false, false);
        acc[1][2] = __builtin_amdgcn_wmma_f32_16x16x32_bf16(false, a1, false, b2, (short)0, acc[1][2], false, false);
        acc[1][3] = __builtin_amdgcn_wmma_f32_16x16x32_bf16(false, a1, false, b3, (short)0, acc[1][3], false, false);
    }
#endif

    // C/D layout: VGPR i -> M = i + 8*(lane>=16), N = lane&15
    const int colw = lane & 15;
    const int mofs = (lane >> 4) << 3;
#pragma unroll
    for (int p = 0; p < 2; ++p) {
#pragma unroll
        for (int t = 0; t < 4; ++t) {
            int n = n0 + t * 16 + colw;
#pragma unroll
            for (int i = 0; i < 8; ++i) {
                int m = m0 + p * 16 + mofs + i;
                float v = acc[p][t][i] * scale;
                size_t idx = TRANS_C ? (coff + (size_t)n * ldc + m)
                                     : (coff + (size_t)m * ldc + n);
                if (OUT_BF16) ((bf16*)Cv)[idx] = (bf16)v;
                else          ((float*)Cv)[idx] = v;
            }
        }
    }
}

// ---------------------------------------------------------------------------
// Masked softmax: logits fp32 [rows, S] -> atten bf16 (zero logits -> -9e15)
// ---------------------------------------------------------------------------
__global__ void __launch_bounds__(256)
softmax_mask_kernel(const float* __restrict__ logits, bf16* __restrict__ atten, int S)
{
    const size_t row = blockIdx.x;
    const float* in = logits + row * (size_t)S;
    bf16* out = atten + row * (size_t)S;
    const int tid = threadIdx.x;
    const int nv = S >> 8;               // S / 256, assumed <= 8

    float vals[8];
    float mx = -3.0e38f;
    for (int i = 0; i < nv; ++i) {
        float v = in[tid + (i << 8)];
        v = (v != 0.0f) ? v : -9.0e15f;
        vals[i] = v;
        mx = fmaxf(mx, v);
    }

    __shared__ float red[256];
    red[tid] = mx; __syncthreads();
    for (int s2 = 128; s2 > 0; s2 >>= 1) {
        if (tid < s2) red[tid] = fmaxf(red[tid], red[tid + s2]);
        __syncthreads();
    }
    mx = red[0]; __syncthreads();

    float sum = 0.0f;
    for (int i = 0; i < nv; ++i) { vals[i] = __expf(vals[i] - mx); sum += vals[i]; }
    red[tid] = sum; __syncthreads();
    for (int s2 = 128; s2 > 0; s2 >>= 1) {
        if (tid < s2) red[tid] += red[tid + s2];
        __syncthreads();
    }
    float inv = 1.0f / red[0];
    for (int i = 0; i < nv; ++i) out[tid + (i << 8)] = (bf16)(vals[i] * inv);
}

// ---------------------------------------------------------------------------
// Orchestration
// ---------------------------------------------------------------------------
extern "C" void kernel_launch(void* const* d_in, const int* in_sizes, int n_in,
                              void* d_out, int out_size, void* d_ws, size_t ws_size,
                              hipStream_t stream)
{
    (void)in_sizes; (void)n_in; (void)out_size; (void)ws_size;

    constexpr int  Bn = 4, S = 1024, E = 1024, H = 8, D = 1024;
    constexpr long F  = (long)H * D;        // 8192

    const float* q  = (const float*)d_in[0];
    const float* k  = (const float*)d_in[1];
    const float* v  = (const float*)d_in[2];
    const float* Wq = (const float*)d_in[3];
    const float* Wk = (const float*)d_in[4];
    const float* Wv = (const float*)d_in[5];
    const float* Wl = (const float*)d_in[6];

    char* ws = (char*)d_ws;
    size_t off = 0;
    auto alloc = [&](size_t bytes) {
        void* p = ws + off;
        off += (bytes + 255) & ~(size_t)255;
        return p;
    };

    const size_t nX  = (size_t)Bn * S * E;
    const size_t nW  = (size_t)H * D * E;
    const size_t nWl = (size_t)E * F;
    const size_t nP  = (size_t)Bn * H * S * (size_t)D;

    bf16* qb   = (bf16*)alloc(nX * 2);
    bf16* kb   = (bf16*)alloc(nX * 2);
    bf16* vb   = (bf16*)alloc(nX * 2);
    bf16* Wqb  = (bf16*)alloc(nW * 2);
    bf16* Wkb  = (bf16*)alloc(nW * 2);
    bf16* Wvb  = (bf16*)alloc(nW * 2);
    bf16* Wlb  = (bf16*)alloc(nWl * 2);
    bf16* inq  = (bf16*)alloc(nP * 2);   // reused as atten after logits
    bf16* ink  = (bf16*)alloc(nP * 2);   // reused as hcat after logits
    bf16* invT = (bf16*)alloc(nP * 2);   // v projection stored transposed: [b,h,D,S]
    float* logits = (float*)alloc((size_t)Bn * H * S * (size_t)S * 4);
    bf16* atten = inq;
    bf16* hcat  = ink;

    f32_to_bf16_kernel<<<2048, 256, 0, stream>>>(q,  qb,  nX);
    f32_to_bf16_kernel<<<2048, 256, 0, stream>>>(k,  kb,  nX);
    f32_to_bf16_kernel<<<2048, 256, 0, stream>>>(v,  vb,  nX);
    f32_to_bf16_kernel<<<4096, 256, 0, stream>>>(Wq, Wqb, nW);
    f32_to_bf16_kernel<<<4096, 256, 0, stream>>>(Wk, Wkb, nW);
    f32_to_bf16_kernel<<<4096, 256, 0, stream>>>(Wv, Wvb, nW);
    f32_to_bf16_kernel<<<4096, 256, 0, stream>>>(Wl, Wlb, nWl);

    dim3 blk(128);
    dim3 gBH(1024 / 64, 1024 / 128, Bn * H);  // 16 x 8 x 32
    dim3 gB (1024 / 64, 1024 / 128, Bn);      // 16 x 8 x 4

    // projections: in_q/in_k = x[b] @ W[h]^T  (M=S, N=D, K=E)
    gemm_nt_wmma<true, false><<<gBH, blk, 0, stream>>>(
        qb, Wqb, inq, E, E, E, D, 3,
        (long)S * E, 0L, 0L, (long)D * E, (long)H * S * D, (long)S * D, 1.0f);
    gemm_nt_wmma<true, false><<<gBH, blk, 0, stream>>>(
        kb, Wkb, ink, E, E, E, D, 3,
        (long)S * E, 0L, 0L, (long)D * E, (long)H * S * D, (long)S * D, 1.0f);
    // in_v stored transposed ([b,h,D,S]) so the PV GEMM is also NT
    gemm_nt_wmma<true, true><<<gBH, blk, 0, stream>>>(
        vb, Wvb, invT, E, E, E, S, 3,
        (long)S * E, 0L, 0L, (long)D * E, (long)H * D * S, (long)D * S, 1.0f);

    // logits = (in_q @ in_k^T) / sqrt(S)
    gemm_nt_wmma<false, false><<<gBH, blk, 0, stream>>>(
        inq, ink, logits, D, D, D, S, 3,
        (long)H * S * D, (long)S * D, (long)H * S * D, (long)S * D,
        (long)H * S * S, (long)S * S, 0.03125f);

    // masked softmax -> bf16 atten
    softmax_mask_kernel<<<Bn * H * S, 256, 0, stream>>>(logits, atten, S);

    // out_h = atten @ in_v, written into hcat[b, s, h*D + d]
    gemm_nt_wmma<true, false><<<gBH, blk, 0, stream>>>(
        atten, invT, hcat, S, S, S, (int)F, 3,
        (long)H * S * S, (long)S * S, (long)H * D * S, (long)D * S,
        (long)S * F, (long)D, 1.0f);

    // out = hcat @ Wlast^T  (M=S, N=E, K=F), fp32 out
    gemm_nt_wmma<false, false><<<gB, blk, 0, stream>>>(
        hcat, Wlb, (float*)d_out, (int)F, (int)F, (int)F, E, 0,
        (long)S * F, 0L, 0L, 0L, (long)S * E, 0L, 1.0f);
}
